// MoveEncoder_35321811042988
// MI455X (gfx1250) — matrix operand order repaired and checked by reading, not compiled, not run on hardware
//
#include <hip/hip_runtime.h>
#include <hip/hip_bf16.h>

typedef __attribute__((ext_vector_type(16))) _Float16 v16h;
typedef __attribute__((ext_vector_type(8)))  _Float16 v8h;
typedef __attribute__((ext_vector_type(8)))  float    v8f;

#define NMOVES 262144
#define DM 192      // d_model
#define HH 384      // inner dim
#define KCOMB 32    // padded [flags(7) | cons(12) | 0...] input dim
#define HCOMB 768   // combined hidden (flag 384 + cons 384)

// d_ws layout (in _Float16 elements), all weights pre-transposed to [Ncol][K]
#define OFF_W1T   0                        // [768][32]  block-diag {flag_W1, cons_W1}
#define OFF_W2T   (OFF_W1T + HCOMB*KCOMB)  // [192][768] stacked {flag_W2; cons_W2}
#define OFF_OW1T  (OFF_W2T + DM*HCOMB)     // [384][192] out_W1^T
#define OFF_OW2T  (OFF_OW1T + HH*DM)       // [192][384] out_W2^T
#define WS_HALF_TOTAL (OFF_OW2T + DM*HH)   // 319488 halves (~624 KB)

__device__ __forceinline__ float gelu_exact(float x) {
  return 0.5f * x * (1.0f + erff(x * 0.7071067811865475f));
}

__device__ __forceinline__ v8f wmma16(v16h a, v16h b, v8f c) {
  // D = A(16x32 f16) * B(32x16 f16) + C(16x16 f32)
  return __builtin_amdgcn_wmma_f32_16x16x32_f16(false, a, false, b, (short)0, c,
                                                false, false);
}

// Per-wave LDS producer->consumer sync. DS instructions from one wave execute
// in order (ISA: DScnt is in-order), so no s_barrier is needed for intra-wave
// cross-lane exchange; we only (a) wait for outstanding DS stores and
// (b) stop the compiler from reordering memory ops across this point.
__device__ __forceinline__ void wave_lds_sync() {
  asm volatile("s_wait_dscnt 0x0" ::: "memory");
}
// Pure compiler fence (hardware in-order DS makes any emitted order correct;
// we just must keep program order of LDS ops intact).
__device__ __forceinline__ void lds_fence() {
  asm volatile("" ::: "memory");
}

// A-fragment: lane holds row M=lane&15; K elements are two contiguous runs of 8
// (callers pass o2 = o1 + 8: k 0..15 for lanes 0-15, 16..31 for lanes 16-31).
__device__ __forceinline__ v16h load_afrag(const _Float16* row, int o1, int o2) {
  v8h lo = *(const v8h*)(row + o1);
  v8h hi = *(const v8h*)(row + o2);
  return __builtin_shufflevector(lo, hi, 0,1,2,3,4,5,6,7,8,9,10,11,12,13,14,15);
}

// B-fragment: lane holds col N=lane&15; K = one contiguous run of 16
// (0..15 for lanes 0-15, 16..31 for lanes 16-31) -> p already includes +16*half.
__device__ __forceinline__ v16h load_bfrag(const _Float16* p) {
  v8h lo = *(const v8h*)(p);
  v8h hi = *(const v8h*)(p + 8);
  return __builtin_shufflevector(lo, hi, 0,1,2,3,4,5,6,7,8,9,10,11,12,13,14,15);
}

// ---------------- pre-pass: fp32 weights -> transposed fp16 in d_ws ----------
__global__ __launch_bounds__(256) void prep_weights(
    const float* __restrict__ fW1, const float* __restrict__ cW1,
    const float* __restrict__ fW2, const float* __restrict__ cW2,
    const float* __restrict__ oW1, const float* __restrict__ oW2,
    _Float16* __restrict__ ws) {
  int i = blockIdx.x * 256 + threadIdx.x;
  if (i < HCOMB * KCOMB) {                      // W1T[n][k]: block-diagonal
    int n = i / KCOMB, k = i % KCOMB;
    float v = 0.f;
    if (n < HH)        { if (k < 7)            v = fW1[k * HH + n]; }
    else               { if (k >= 7 && k < 19) v = cW1[(k - 7) * HH + (n - HH)]; }
    ws[OFF_W1T + i] = (_Float16)v;
    return;
  }
  i -= HCOMB * KCOMB;
  if (i < DM * HCOMB) {                         // W2T[n][k]: stacked {flag;cons}
    int n = i / HCOMB, k = i % HCOMB;
    float v = (k < HH) ? fW2[k * DM + n] : cW2[(k - HH) * DM + n];
    ws[OFF_W2T + i] = (_Float16)v;
    return;
  }
  i -= DM * HCOMB;
  if (i < HH * DM) {                            // OW1T[n][k] = out_W1[k][n]
    int n = i / DM, k = i % DM;
    ws[OFF_OW1T + i] = (_Float16)oW1[k * HH + n];
    return;
  }
  i -= HH * DM;
  if (i < DM * HH) {                            // OW2T[n][k] = out_W2[k][n]
    int n = i / HH, k = i % HH;
    ws[OFF_OW2T + i] = (_Float16)oW2[k * DM + n];
  }
}

// ---------------- main fused kernel: 8 waves/WG, 16 moves per wave -----------
__global__ __launch_bounds__(256) void move_encoder_kernel(
    const float* __restrict__ flags,  const float* __restrict__ cons,
    const int* __restrict__ from_sq,  const int* __restrict__ to_sq,
    const int* __restrict__ kind_id,  const int* __restrict__ promo_id,
    const int* __restrict__ meta_id,  const int* __restrict__ mover_idx,
    const int* __restrict__ target_idx,
    const float* __restrict__ gctx,   const float* __restrict__ piece,
    const float* __restrict__ sqe,    const float* __restrict__ kinde,
    const float* __restrict__ promoe, const float* __restrict__ metae,
    const float* __restrict__ flag_b1, const float* __restrict__ flag_b2,
    const float* __restrict__ cons_b1, const float* __restrict__ cons_b2,
    const float* __restrict__ ln_g,    const float* __restrict__ ln_b,
    const float* __restrict__ out_b1,  const float* __restrict__ out_b2,
    const _Float16* __restrict__ ws,   float* __restrict__ out) {
  __shared__ __align__(16) _Float16 xbuf[8][16][DM];  // normalized tokens (f16)
  __shared__ __align__(16) _Float16 hbuf[8][16][32];  // per-wave staging chunk
  __shared__ int idxs[8][16][8];

  const int tid  = threadIdx.x;
  const int w    = tid >> 5;        // wave id in WG
  const int lane = tid & 31;
  const int half = lane >> 4;       // 0: lanes 0-15, 1: lanes 16-31
  const int c    = lane & 15;       // column / A-row index
  const int waveBase = blockIdx.x * 128 + w * 16;

  const _Float16* W1T  = ws + OFF_W1T;
  const _Float16* W2T  = ws + OFF_W2T;
  const _Float16* OW1T = ws + OFF_OW1T;
  const _Float16* OW2T = ws + OFF_OW2T;

  // ---- stage this wave's 7 index streams into LDS (wave-local) ------------
  if (lane < 16) {
    const int m = waveBase + lane;
    idxs[w][lane][0] = mover_idx[m];
    idxs[w][lane][1] = target_idx[m];
    idxs[w][lane][2] = from_sq[m];
    idxs[w][lane][3] = to_sq[m];
    idxs[w][lane][4] = kind_id[m];
    idxs[w][lane][5] = promo_id[m];
    idxs[w][lane][6] = meta_id[m];
  }
  wave_lds_sync();

  // ---- gather embeddings + fused flag_b2/cons_b2 directly into C layout ---
  // C layout: lane(0-15)->M=r, lane(16-31)->M=r+8; VGPR r; column N = t*16+c.
  v8f tokC[12];
  #pragma unroll
  for (int t = 0; t < 12; ++t) {
    const int d = t * 16 + c;
    const float base = gctx[d] + flag_b2[d] + cons_b2[d];
    #pragma unroll
    for (int r = 0; r < 8; ++r) tokC[t][r] = base;
  }
  #pragma unroll
  for (int r = 0; r < 8; ++r) {
    const int M = r + 8 * half;
    const int* ix = idxs[w][M];
    const float* p0 = piece  + ix[0] * DM + c;
    const float* p1 = piece  + ix[1] * DM + c;
    const float* p2 = sqe    + ix[2] * DM + c;
    const float* p3 = sqe    + ix[3] * DM + c;
    const float* p4 = kinde  + ix[4] * DM + c;
    const float* p5 = promoe + ix[5] * DM + c;
    const float* p6 = metae  + ix[6] * DM + c;
    #pragma unroll
    for (int t = 0; t < 12; ++t) {
      const int o = t * 16;   // immediate-offset loads off 7 base pointers
      tokC[t][r] += ((p0[o] + p1[o]) + (p2[o] + p3[o])) +
                    ((p4[o] + p5[o]) + p6[o]);
    }
  }

  // ---- stage padded [flags|cons|0] (16x32 f16) into hbuf, coalesced -------
  v16h aIn;
  {
    v8h z = {};
    *(v8h*)(&hbuf[w][c][16 * half])     = z;     // zero the 16x32 pad region
    *(v8h*)(&hbuf[w][c][16 * half + 8]) = z;
    lds_fence();
    const int fbase = waveBase * 7;              // 112 contiguous floats
    #pragma unroll
    for (int j = 0; j < 4; ++j) {
      const int i = lane + 32 * j;
      if (i < 112) hbuf[w][i / 7][i % 7] = (_Float16)flags[fbase + i];
    }
    const int cbase = waveBase * 12;             // 192 contiguous floats
    #pragma unroll
    for (int j = 0; j < 6; ++j) {
      const int i = lane + 32 * j;
      hbuf[w][i / 12][7 + i % 12] = (_Float16)cons[cbase + i];
    }
    wave_lds_sync();
    aIn = load_afrag(&hbuf[w][c][0], 16 * half, 16 * half + 8);
    lds_fence();
  }

  // ---- fused GEMM1 (K=32) -> GELU -> GEMM2 (K=768, accumulate into tokC) --
  for (int kc = 0; kc < 24; ++kc) {
    const int n0 = kc * 32;
    const _Float16* pB = W2T + (size_t)c * HCOMB + n0 + 16 * half;

    // ALL 12 GEMM2 B tiles issued before the GELU/LDS phase (and before the
    // dscnt-sync memory clobber) so they are in flight across it.
    v16h B0[6], B1[6];
    #pragma unroll
    for (int j = 0; j < 6; ++j)
      B0[j] = load_bfrag(pB + (size_t)j * 16 * HCOMB);

    v8f h0 = {}, h1 = {};
    {
      v16h b0 = load_bfrag(W1T + (size_t)(n0 + c)      * KCOMB + 16 * half);
      v16h b1 = load_bfrag(W1T + (size_t)(n0 + 16 + c) * KCOMB + 16 * half);
      h0 = wmma16(aIn, b0, h0);
      h1 = wmma16(aIn, b1, h1);
    }
    #pragma unroll
    for (int j = 0; j < 6; ++j)
      B1[j] = load_bfrag(pB + (size_t)(j + 6) * 16 * HCOMB);

    float bias0, bias1;
    if (n0 < HH) { bias0 = flag_b1[n0 + c];      bias1 = flag_b1[n0 + 16 + c]; }
    else         { bias0 = cons_b1[n0 - HH + c]; bias1 = cons_b1[n0 - HH + 16 + c]; }
    #pragma unroll
    for (int r = 0; r < 8; ++r) {
      const int M = r + 8 * half;
      hbuf[w][M][c]      = (_Float16)gelu_exact(h0[r] + bias0);
      hbuf[w][M][16 + c] = (_Float16)gelu_exact(h1[r] + bias1);
    }
    wave_lds_sync();
    v16h aH = load_afrag(&hbuf[w][c][0], 16 * half, 16 * half + 8);

    #pragma unroll
    for (int j = 0; j < 6; ++j) tokC[j]     = wmma16(aH, B0[j], tokC[j]);
    #pragma unroll
    for (int j = 0; j < 6; ++j) tokC[j + 6] = wmma16(aH, B1[j], tokC[j + 6]);
    lds_fence();   // keep next iteration's hbuf stores after this aH load
  }

  // ---- LayerNorm over D=192, in registers + shfl butterflies --------------
  float mu[8], rstd[8];
  #pragma unroll
  for (int r = 0; r < 8; ++r) {
    float s = 0.f, q = 0.f;
    #pragma unroll
    for (int t = 0; t < 12; ++t) { const float v = tokC[t][r]; s += v; q += v * v; }
    #pragma unroll
    for (int off = 1; off < 16; off <<= 1) {   // reduce across the 16-col group
      s += __shfl_xor(s, off, 32);
      q += __shfl_xor(q, off, 32);
    }
    const float m = s * (1.0f / 192.0f);
    mu[r]   = m;
    rstd[r] = rsqrtf(q * (1.0f / 192.0f) - m * m + 1e-5f);
  }
  #pragma unroll
  for (int t = 0; t < 12; ++t) {
    const int d = t * 16 + c;
    const float g = ln_g[d], b = ln_b[d];
    #pragma unroll
    for (int r = 0; r < 8; ++r) {
      const int M = r + 8 * half;
      xbuf[w][M][d] = (_Float16)((tokC[t][r] - mu[r]) * rstd[r] * g + b);
    }
  }
  wave_lds_sync();

  // x as A fragments for K=192 (6 chunks of 32)
  v16h xA[6];
  #pragma unroll
  for (int kk = 0; kk < 6; ++kk)
    xA[kk] = load_afrag(&xbuf[w][c][0], kk * 32 + 16 * half, kk * 32 + 16 * half + 8);
  lds_fence();

  // reuse tokC as output accumulator
  #pragma unroll
  for (int t = 0; t < 12; ++t) { v8f zz = {}; tokC[t] = zz; }

  // ---- fused GEMM3 (K=192) -> GELU -> GEMM4 (K=384, accumulate) -----------
  for (int kc = 0; kc < 12; ++kc) {
    const int n0 = kc * 32;
    const _Float16* pB = OW2T + (size_t)c * HH + n0 + 16 * half;

    // batch 0 of GEMM4 B tiles in flight across GEMM3
    v16h B0[6], B1[6];
    #pragma unroll
    for (int j = 0; j < 6; ++j)
      B0[j] = load_bfrag(pB + (size_t)j * 16 * HH);

    v8f u0 = {}, u1 = {};
    {
      const _Float16* pU0 = OW1T + (size_t)(n0 + c)      * DM + 16 * half;
      const _Float16* pU1 = OW1T + (size_t)(n0 + 16 + c) * DM + 16 * half;
      v16h BU[12];                       // all 12 GEMM3 fragments in flight
      #pragma unroll
      for (int kk = 0; kk < 6; ++kk) {
        BU[2 * kk]     = load_bfrag(pU0 + kk * 32);
        BU[2 * kk + 1] = load_bfrag(pU1 + kk * 32);
      }
      #pragma unroll
      for (int kk = 0; kk < 6; ++kk) {
        u0 = wmma16(xA[kk], BU[2 * kk],     u0);
        u1 = wmma16(xA[kk], BU[2 * kk + 1], u1);
      }
    }
    // batch 1 issued before the GELU/LDS phase so it overlaps it
    #pragma unroll
    for (int j = 0; j < 6; ++j)
      B1[j] = load_bfrag(pB + (size_t)(j + 6) * 16 * HH);

    const float bias0 = out_b1[n0 + c];
    const float bias1 = out_b1[n0 + 16 + c];
    #pragma unroll
    for (int r = 0; r < 8; ++r) {
      const int M = r + 8 * half;
      hbuf[w][M][c]      = (_Float16)gelu_exact(u0[r] + bias0);
      hbuf[w][M][16 + c] = (_Float16)gelu_exact(u1[r] + bias1);
    }
    wave_lds_sync();
    v16h aU = load_afrag(&hbuf[w][c][0], 16 * half, 16 * half + 8);

    #pragma unroll
    for (int j = 0; j < 6; ++j) tokC[j]     = wmma16(aU, B0[j], tokC[j]);
    #pragma unroll
    for (int j = 0; j < 6; ++j) tokC[j + 6] = wmma16(aU, B1[j], tokC[j + 6]);
    lds_fence();
  }

  // ---- +out_b2, coalesced fp32 store --------------------------------------
  #pragma unroll
  for (int t = 0; t < 12; ++t) {
    const int d = t * 16 + c;
    const float ob = out_b2[d];
    #pragma unroll
    for (int r = 0; r < 8; ++r) {
      const int M = r + 8 * half;
      out[(size_t)(waveBase + M) * DM + d] = tokC[t][r] + ob;
    }
  }
}

extern "C" void kernel_launch(void* const* d_in, const int* in_sizes, int n_in,
                              void* d_out, int out_size, void* d_ws, size_t ws_size,
                              hipStream_t stream) {
  const float* flags      = (const float*)d_in[0];
  const float* cons       = (const float*)d_in[1];
  const int*   from_sq    = (const int*)d_in[2];
  const int*   to_sq      = (const int*)d_in[3];
  const int*   kind_id    = (const int*)d_in[4];
  const int*   promo_id   = (const int*)d_in[5];
  const int*   meta_id    = (const int*)d_in[6];
  const int*   mover_idx  = (const int*)d_in[7];
  const int*   target_idx = (const int*)d_in[8];
  const float* gctx       = (const float*)d_in[9];
  const float* piece      = (const float*)d_in[10];
  const float* sqe        = (const float*)d_in[11];
  const float* kinde      = (const float*)d_in[12];
  const float* promoe     = (const float*)d_in[13];
  const float* metae      = (const float*)d_in[14];
  const float* flag_W1    = (const float*)d_in[15];
  const float* flag_b1    = (const float*)d_in[16];
  const float* flag_W2    = (const float*)d_in[17];
  const float* flag_b2    = (const float*)d_in[18];
  const float* cons_W1    = (const float*)d_in[19];
  const float* cons_b1    = (const float*)d_in[20];
  const float* cons_W2    = (const float*)d_in[21];
  const float* cons_b2    = (const float*)d_in[22];
  const float* ln_g       = (const float*)d_in[23];
  const float* ln_b       = (const float*)d_in[24];
  const float* out_W1     = (const float*)d_in[25];
  const float* out_b1     = (const float*)d_in[26];
  const float* out_W2     = (const float*)d_in[27];
  const float* out_b2     = (const float*)d_in[28];
  float* out = (float*)d_out;
  _Float16* ws = (_Float16*)d_ws;

  prep_weights<<<(WS_HALF_TOTAL + 255) / 256, 256, 0, stream>>>(
      flag_W1, cons_W1, flag_W2, cons_W2, out_W1, out_W2, ws);

  move_encoder_kernel<<<NMOVES / 128, 256, 0, stream>>>(
      flags, cons, from_sq, to_sq, kind_id, promo_id, meta_id, mover_idx,
      target_idx, gctx, piece, sqe, kinde, promoe, metae,
      flag_b1, flag_b2, cons_b1, cons_b2, ln_g, ln_b, out_b1, out_b2,
      ws, out);
}
